// MultiHeadSelfAttention_72499047956752
// MI455X (gfx1250) — matrix operand order repaired and checked by reading, not compiled, not run on hardware
//
#include <hip/hip_runtime.h>
#include <hip/hip_bf16.h>

// ---------------------------------------------------------------------------
// MultiHeadSelfAttention for MI455X (gfx1250), bf16 WMMA + flash attention.
// B=4, S=2048, D=1024, H=16, DK=64
// ---------------------------------------------------------------------------

#define B_  4
#define S_  2048
#define D_  1024
#define H_  16
#define DK_ 64

typedef __bf16 bf16_t;
typedef __attribute__((ext_vector_type(16))) __bf16 v16bf;
typedef __attribute__((ext_vector_type(8)))  __bf16 v8bf;
typedef __attribute__((ext_vector_type(8)))  float  v8f;

static __device__ __forceinline__ bf16_t f2bf(float f) {
    union { float f; unsigned u; } v; v.f = f;
    unsigned r = v.u + 0x7FFFu + ((v.u >> 16) & 1u);   // round to nearest even
    unsigned short h = (unsigned short)(r >> 16);
    return __builtin_bit_cast(bf16_t, h);
}

static __device__ __forceinline__ v8f zero8() {
    v8f z;
#pragma unroll
    for (int i = 0; i < 8; ++i) z[i] = 0.0f;
    return z;
}

// A fragment: 16x32 bf16, row-major source, row stride ld (elements).
// ISA layout: lane<16 -> row=lane, K = {0..7, 16..23}; lane>=16 -> K={8..15,24..31}.
static __device__ __forceinline__ v16bf frag_a_rm(const bf16_t* src, int ld) {
    int lane = threadIdx.x & 31;
    int r = lane & 15, hi = lane >> 4;
    const bf16_t* p = src + r * ld + hi * 8;
    v8bf lo  = *(const v8bf*)(p);
    v8bf hi8 = *(const v8bf*)(p + 16);
    v16bf a;
#pragma unroll
    for (int i = 0; i < 8; ++i) { a[i] = lo[i]; a[8 + i] = hi8[i]; }
    return a;
}

// B fragment: 32x16 bf16 where B[kk][n] = srcT[n*ld + kk]  (srcT = B transposed,
// row-major with stride ld). Lane<16 holds column n=lane, K={0..7,16..23}; lane>=16 K={8..15,24..31}.
static __device__ __forceinline__ v16bf frag_b_cm(const bf16_t* srcT, int ld) {
    int lane = threadIdx.x & 31;
    int n = lane & 15, hi = lane >> 4;
    const bf16_t* p = srcT + n * ld + hi * 8;
    v8bf b0 = *(const v8bf*)(p);
    v8bf b1 = *(const v8bf*)(p + 16);
    v16bf b;
#pragma unroll
    for (int i = 0; i < 8; ++i) { b[i] = b0[i]; b[8 + i] = b1[i]; }
    return b;
}

static __device__ __forceinline__ v8f wmma_bf16(v16bf a, v16bf b, v8f c) {
    return __builtin_amdgcn_wmma_f32_16x16x32_bf16(
        /*neg_a=*/false, a, /*neg_b=*/false, b,
        /*c_mod=*/(short)0, c, /*reuse_a=*/false, /*reuse_b=*/false);
}

static __device__ __forceinline__ float redmax16(float v) {
    v = fmaxf(v, __shfl_xor(v, 1, 16));
    v = fmaxf(v, __shfl_xor(v, 2, 16));
    v = fmaxf(v, __shfl_xor(v, 4, 16));
    v = fmaxf(v, __shfl_xor(v, 8, 16));
    return v;
}
static __device__ __forceinline__ float redsum16(float v) {
    v += __shfl_xor(v, 1, 16);
    v += __shfl_xor(v, 2, 16);
    v += __shfl_xor(v, 4, 16);
    v += __shfl_xor(v, 8, 16);
    return v;
}

// ---------------------------------------------------------------------------
// Kernel 1: weight transpose + fp32 -> bf16.  Wt[n][k] = (bf16)W[k][n]
// grid (4096, 4), block 256
// ---------------------------------------------------------------------------
__global__ void wt_kernel(const float* __restrict__ W0, const float* __restrict__ W1,
                          const float* __restrict__ W2, const float* __restrict__ W3,
                          bf16_t* __restrict__ T0, bf16_t* __restrict__ T1,
                          bf16_t* __restrict__ T2, bf16_t* __restrict__ T3) {
    const float* W = (blockIdx.y == 0) ? W0 : (blockIdx.y == 1) ? W1
                    : (blockIdx.y == 2) ? W2 : W3;
    bf16_t* T = (blockIdx.y == 0) ? T0 : (blockIdx.y == 1) ? T1
               : (blockIdx.y == 2) ? T2 : T3;
    int idx = blockIdx.x * 256 + threadIdx.x;       // 0 .. 1M-1
    int k = idx & (D_ - 1);
    int n = idx >> 10;
    T[idx] = f2bf(W[k * D_ + n]);                   // T[n*1024 + k]
}

// ---------------------------------------------------------------------------
// Kernel 2: Q/K/V projections.  C = x(bf16) @ W, 128x128 tile, 8 waves.
// z = 0:Q  1:K  2:V.  Q,K stored [B,H,S,DK]; V stored transposed [B,H,DK,S].
// Head mapping (faithful to reference reshape): h = n & 15, dk = n >> 4.
// grid (64, 8, 3), block 256
// ---------------------------------------------------------------------------
__global__ __launch_bounds__(256)
void proj_kernel(const float* __restrict__ x,
                 const bf16_t* __restrict__ WQt, const bf16_t* __restrict__ WKt,
                 const bf16_t* __restrict__ WVt,
                 bf16_t* __restrict__ Qb, bf16_t* __restrict__ Kb,
                 bf16_t* __restrict__ Vtb) {
    __shared__ bf16_t ldsA[128 * 32];   // [m][k]
    __shared__ bf16_t ldsB[128 * 32];   // [n][k]  (W transposed)

    const int z  = blockIdx.z;
    const bf16_t* Wt = (z == 0) ? WQt : (z == 1) ? WKt : WVt;
    const int m0 = blockIdx.x * 128;
    const int n0 = blockIdx.y * 128;
    const int tid  = threadIdx.x;
    const int w    = tid >> 5;
    const int lane = tid & 31;
    const int wr = w >> 1;     // 0..3 : rows 32*wr
    const int wc = w & 1;      // 0..1 : cols 64*wc

    v8f acc[2][4];
#pragma unroll
    for (int i = 0; i < 2; ++i)
#pragma unroll
        for (int j = 0; j < 4; ++j) acc[i][j] = zero8();

    for (int k0 = 0; k0 < D_; k0 += 32) {
        __syncthreads();
        {   // stage A: 128x32 of x, convert fp32 -> bf16
            int r = tid >> 1, c0 = (tid & 1) * 16;
            const float* src = x + (m0 + r) * D_ + k0 + c0;
            bf16_t* dst = ldsA + r * 32 + c0;
#pragma unroll
            for (int i = 0; i < 16; ++i) dst[i] = f2bf(src[i]);
        }
        {   // stage B: 128x32 of Wt (already bf16, contiguous along k)
            int r = tid >> 1, c0 = (tid & 1) * 16;
            const bf16_t* src = Wt + (n0 + r) * D_ + k0 + c0;
            *(v8bf*)(ldsB + r * 32 + c0)     = *(const v8bf*)(src);
            *(v8bf*)(ldsB + r * 32 + c0 + 8) = *(const v8bf*)(src + 8);
        }
        __syncthreads();

        v16bf a0 = frag_a_rm(ldsA + (wr * 32 + 0)  * 32, 32);
        v16bf a1 = frag_a_rm(ldsA + (wr * 32 + 16) * 32, 32);
#pragma unroll
        for (int j = 0; j < 4; ++j) {
            v16bf b = frag_b_cm(ldsB + (wc * 64 + j * 16) * 32, 32);
            acc[0][j] = wmma_bf16(a0, b, acc[0][j]);
            acc[1][j] = wmma_bf16(a1, b, acc[1][j]);
        }
    }

    // epilogue: scatter into head layout
    const int nL = lane & 15, hiL = lane >> 4;
#pragma unroll
    for (int i = 0; i < 2; ++i) {
#pragma unroll
        for (int j = 0; j < 4; ++j) {
            int mbase = m0 + wr * 32 + i * 16;
            int nglob = n0 + wc * 64 + j * 16 + nL;
            int h  = nglob & (H_ - 1);
            int dk = nglob >> 4;
#pragma unroll
            for (int v = 0; v < 8; ++v) {
                int m = mbase + v + 8 * hiL;
                int b = m >> 11;               // /S_
                int s = m & (S_ - 1);
                bf16_t val = f2bf(acc[i][j][v]);
                if (z == 2)      Vtb[((b * H_ + h) * DK_ + dk) * S_ + s]  = val;
                else if (z == 1) Kb [((b * H_ + h) * S_  + s) * DK_ + dk] = val;
                else             Qb [((b * H_ + h) * S_  + s) * DK_ + dk] = val;
            }
        }
    }
}

// ---------------------------------------------------------------------------
// Kernel 3: flash attention.  grid (S/128=16, B*H=64), block 256 (8 waves).
// Each wave owns 16 query rows; 64-key chunks of K/V staged in LDS.
// attn output (pre-WO) stored bf16 [B,S,D] with feature = h*64 + d.
// ---------------------------------------------------------------------------
__global__ __launch_bounds__(256)
void flash_kernel(const bf16_t* __restrict__ Qb, const bf16_t* __restrict__ Kb,
                  const bf16_t* __restrict__ Vtb,
                  const unsigned char* __restrict__ mask,
                  bf16_t* __restrict__ attn) {
    __shared__ bf16_t ldsK[64 * 64];      // [key][d]
    __shared__ bf16_t ldsV[64 * 64];      // [d][key]   (from V transposed)
    __shared__ float  maskAdd[64];
    __shared__ bf16_t ldsP[8][16 * 64];   // per-wave P transpose scratch

    const int bh = blockIdx.y;            // 0..63
    const int b  = bh >> 4, h = bh & (H_ - 1);
    const int w    = threadIdx.x >> 5;
    const int lane = threadIdx.x & 31;
    const int q0 = blockIdx.x * 128 + w * 16;

    const bf16_t* Qhead = Qb  + (bh * S_ + q0) * DK_;
    const bf16_t* Khead = Kb  + (size_t)bh * S_ * DK_;
    const bf16_t* Vhead = Vtb + (size_t)bh * DK_ * S_;

    const v16bf qa0 = frag_a_rm(Qhead, DK_);        // d = 0..31
    const v16bf qa1 = frag_a_rm(Qhead + 32, DK_);   // d = 32..63

    float m_r[8], l_r[8];
    v8f o[4];
#pragma unroll
    for (int v = 0; v < 8; ++v) { m_r[v] = -3.0e38f; l_r[v] = 0.0f; }
#pragma unroll
    for (int t = 0; t < 4; ++t) o[t] = zero8();

    const int nL = lane & 15, hiL = lane >> 4;

    for (int kb = 0; kb < S_; kb += 64) {
        __syncthreads();   // previous chunk's LDS reads complete
        {   // K chunk: contiguous 64x64 bf16 = 512 uint4
            const uint4* src = (const uint4*)(Khead + kb * DK_);
            uint4* dst = (uint4*)ldsK;
            for (int i = threadIdx.x; i < 512; i += 256) dst[i] = src[i];
        }
        {   // V chunk: 64 d-rows (stride S_) x 64 keys
            int d = threadIdx.x >> 2, seg = threadIdx.x & 3;
            const uint4* src = (const uint4*)(Vhead + d * S_ + kb + seg * 16);
            uint4* dst = (uint4*)(ldsV + d * 64 + seg * 16);
            dst[0] = src[0];
            dst[1] = src[1];
        }
        if (threadIdx.x < 64)
            maskAdd[threadIdx.x] =
                mask[b * S_ + kb + threadIdx.x] ? 0.0f : -1.0e6f;
        __syncthreads();

        // ---- scores: 16 (q) x 64 (keys), fp32 accum ----
        v8f sc[4];
#pragma unroll
        for (int t = 0; t < 4; ++t) {
            sc[t] = zero8();
            v16bf bk0 = frag_b_cm(ldsK + (t * 16) * 64, 64);       // d 0..31
            v16bf bk1 = frag_b_cm(ldsK + (t * 16) * 64 + 32, 64);  // d 32..63
            sc[t] = wmma_bf16(qa0, bk0, sc[t]);
            sc[t] = wmma_bf16(qa1, bk1, sc[t]);
        }
        float madd[4];
#pragma unroll
        for (int t = 0; t < 4; ++t) madd[t] = maskAdd[t * 16 + nL];
#pragma unroll
        for (int t = 0; t < 4; ++t)
#pragma unroll
            for (int v = 0; v < 8; ++v)
                sc[t][v] = sc[t][v] * 0.125f + madd[t];   // 1/sqrt(64), key mask

        // ---- online softmax update (row = v + 8*hi per half-wave) ----
        float alpha[8], mnew[8];
#pragma unroll
        for (int v = 0; v < 8; ++v) {
            float mv = sc[0][v];
            mv = fmaxf(mv, sc[1][v]);
            mv = fmaxf(mv, sc[2][v]);
            mv = fmaxf(mv, sc[3][v]);
            mv = redmax16(mv);
            mnew[v]  = fmaxf(m_r[v], mv);
            alpha[v] = __expf(m_r[v] - mnew[v]);
            m_r[v]   = mnew[v];
        }
#pragma unroll
        for (int v = 0; v < 8; ++v) {
            float rs = 0.0f;
#pragma unroll
            for (int t = 0; t < 4; ++t) {
                float p = __expf(sc[t][v] - mnew[v]);
                sc[t][v] = p;
                rs += p;
            }
            rs = redsum16(rs);
            l_r[v] = l_r[v] * alpha[v] + rs;
        }
#pragma unroll
        for (int t = 0; t < 4; ++t)
#pragma unroll
            for (int v = 0; v < 8; ++v) o[t][v] *= alpha[v];

        // ---- P (16x64) -> A-fragment layout via per-wave LDS transpose ----
        bf16_t* P = ldsP[w];
#pragma unroll
        for (int t = 0; t < 4; ++t)
#pragma unroll
            for (int v = 0; v < 8; ++v)
                P[(v + 8 * hiL) * 64 + t * 16 + nL] = f2bf(sc[t][v]);
        asm volatile("" ::: "memory");    // same-wave LDS is in-order in HW
        v16bf pa0 = frag_a_rm(P, 64);          // keys 0..31
        v16bf pa1 = frag_a_rm(P + 32, 64);     // keys 32..63

        // ---- O += P @ V ----
#pragma unroll
        for (int td = 0; td < 4; ++td) {
            v16bf bv0 = frag_b_cm(ldsV + (td * 16) * 64, 64);       // keys 0..31
            v16bf bv1 = frag_b_cm(ldsV + (td * 16) * 64 + 32, 64);  // keys 32..63
            o[td] = wmma_bf16(pa0, bv0, o[td]);
            o[td] = wmma_bf16(pa1, bv1, o[td]);
        }
    }

    // ---- normalize and store attn[b, s, h*64 + d] (bf16) ----
#pragma unroll
    for (int td = 0; td < 4; ++td)
#pragma unroll
        for (int v = 0; v < 8; ++v) {
            int s = q0 + v + 8 * hiL;
            float val = o[td][v] / l_r[v];
            attn[(b * S_ + s) * D_ + h * DK_ + td * 16 + nL] = f2bf(val);
        }
}

// ---------------------------------------------------------------------------
// Kernel 4: output projection + query-mask + abs.  out = |(attn @ WO) * mask|
// grid (64, 8), block 256.  Same tiling as proj_kernel, fp32 output.
// ---------------------------------------------------------------------------
__global__ __launch_bounds__(256)
void oproj_kernel(const bf16_t* __restrict__ attn, const bf16_t* __restrict__ WOt,
                  const unsigned char* __restrict__ mask, float* __restrict__ out) {
    __shared__ bf16_t ldsA[128 * 32];
    __shared__ bf16_t ldsB[128 * 32];

    const int m0 = blockIdx.x * 128;
    const int n0 = blockIdx.y * 128;
    const int tid  = threadIdx.x;
    const int w    = tid >> 5;
    const int lane = tid & 31;
    const int wr = w >> 1, wc = w & 1;

    v8f acc[2][4];
#pragma unroll
    for (int i = 0; i < 2; ++i)
#pragma unroll
        for (int j = 0; j < 4; ++j) acc[i][j] = zero8();

    for (int k0 = 0; k0 < D_; k0 += 32) {
        __syncthreads();
        {
            int r = tid >> 1, c0 = (tid & 1) * 16;
            const bf16_t* src = attn + (m0 + r) * D_ + k0 + c0;
            *(v8bf*)(ldsA + r * 32 + c0)     = *(const v8bf*)(src);
            *(v8bf*)(ldsA + r * 32 + c0 + 8) = *(const v8bf*)(src + 8);
        }
        {
            int r = tid >> 1, c0 = (tid & 1) * 16;
            const bf16_t* src = WOt + (n0 + r) * D_ + k0 + c0;
            *(v8bf*)(ldsB + r * 32 + c0)     = *(const v8bf*)(src);
            *(v8bf*)(ldsB + r * 32 + c0 + 8) = *(const v8bf*)(src + 8);
        }
        __syncthreads();

        v16bf a0 = frag_a_rm(ldsA + (wr * 32 + 0)  * 32, 32);
        v16bf a1 = frag_a_rm(ldsA + (wr * 32 + 16) * 32, 32);
#pragma unroll
        for (int j = 0; j < 4; ++j) {
            v16bf b = frag_b_cm(ldsB + (wc * 64 + j * 16) * 32, 32);
            acc[0][j] = wmma_bf16(a0, b, acc[0][j]);
            acc[1][j] = wmma_bf16(a1, b, acc[1][j]);
        }
    }

    const int nL = lane & 15, hiL = lane >> 4;
#pragma unroll
    for (int i = 0; i < 2; ++i) {
#pragma unroll
        for (int j = 0; j < 4; ++j) {
            int mbase = m0 + wr * 32 + i * 16;
            int nglob = n0 + wc * 64 + j * 16 + nL;
#pragma unroll
            for (int v = 0; v < 8; ++v) {
                int m = mbase + v + 8 * hiL;
                int b = m >> 11;
                int s = m & (S_ - 1);
                float mq = mask[b * S_ + s] ? 1.0f : 0.0f;
                out[m * D_ + nglob] = fabsf(acc[i][j][v] * mq);
            }
        }
    }
}

// ---------------------------------------------------------------------------
// launch
// ---------------------------------------------------------------------------
extern "C" void kernel_launch(void* const* d_in, const int* in_sizes, int n_in,
                              void* d_out, int out_size, void* d_ws, size_t ws_size,
                              hipStream_t stream) {
    const float*         x    = (const float*)d_in[0];
    const unsigned char* mask = (const unsigned char*)d_in[1];   // jnp bool = 1 byte
    const float* WQ = (const float*)d_in[2];
    const float* WK = (const float*)d_in[3];
    const float* WV = (const float*)d_in[4];
    const float* WO = (const float*)d_in[5];
    float* out = (float*)d_out;

    // workspace carve-up (all sizes 256B-aligned already)
    char* ws = (char*)d_ws;
    const size_t QKV_BYTES = (size_t)B_ * H_ * S_ * DK_ * sizeof(bf16_t);  // 16 MB
    const size_t W_BYTES   = (size_t)D_ * D_ * sizeof(bf16_t);             // 2 MB
    bf16_t* Qb   = (bf16_t*)(ws);
    bf16_t* Kb   = (bf16_t*)(ws + QKV_BYTES);
    bf16_t* Vtb  = (bf16_t*)(ws + 2 * QKV_BYTES);
    bf16_t* attn = (bf16_t*)(ws + 3 * QKV_BYTES);
    bf16_t* WQt  = (bf16_t*)(ws + 4 * QKV_BYTES);
    bf16_t* WKt  = (bf16_t*)(ws + 4 * QKV_BYTES + 1 * W_BYTES);
    bf16_t* WVt  = (bf16_t*)(ws + 4 * QKV_BYTES + 2 * W_BYTES);
    bf16_t* WOt  = (bf16_t*)(ws + 4 * QKV_BYTES + 3 * W_BYTES);

    // 1) weights: transpose + convert to bf16
    wt_kernel<<<dim3(D_ * D_ / 256, 4), 256, 0, stream>>>(WQ, WK, WV, WO,
                                                          WQt, WKt, WVt, WOt);
    // 2) Q/K/V projections (z = 0,1,2)
    proj_kernel<<<dim3((B_ * S_) / 128, D_ / 128, 3), 256, 0, stream>>>(
        x, WQt, WKt, WVt, Qb, Kb, Vtb);
    // 3) flash attention
    flash_kernel<<<dim3(S_ / 128, B_ * H_), 256, 0, stream>>>(Qb, Kb, Vtb, mask, attn);
    // 4) output projection + mask + abs
    oproj_kernel<<<dim3((B_ * S_) / 128, D_ / 128), 256, 0, stream>>>(attn, WOt, mask, out);
}
